// LSTMLikeModel_83726092468536
// MI455X (gfx1250) — compile-verified, matrix operand order checked by reading, not run on hardware
//
#include <hip/hip_runtime.h>
#include <hip/hip_bf16.h>

typedef __attribute__((ext_vector_type(16))) __bf16 v16bf;
typedef __attribute__((ext_vector_type(8)))  __bf16 v8bf;
typedef __attribute__((ext_vector_type(8)))  float  v8f;

#define B_   256
#define TIN  256
#define FIN  64
#define UE   512   // encoder units
#define UD   256   // decoder units (per direction)
#define TOUT 64

__device__ __forceinline__ float sigmoidf_(float x) { return 1.0f / (1.0f + expf(-x)); }

// A-fragment loader: for 16-bit WMMA the per-lane A data is two contiguous
// 8-element runs inside the 32-wide K chunk: k = half*8 + [0,8) and
// k = half*8 + 16 + [0,8).  Two aligned 16B loads + shuffle -> v16bf.
__device__ __forceinline__ v16bf load_a_frag(const __bf16* __restrict__ rowk, int half)
{
    const __bf16* p = rowk + (half << 3);
    const v8bf lo = *(const v8bf*)(p);
    const v8bf hi = *(const v8bf*)(p + 16);
    return __builtin_shufflevector(lo, hi, 0, 1, 2, 3, 4, 5, 6, 7,
                                           8, 9, 10, 11, 12, 13, 14, 15);
}

// ---------------------------------------------------------------------------
// Pack a row-major f32 B matrix [K x N] into bf16 WMMA B-fragment order:
//   out[((tile_n * (K/32) + k_chunk) * 32 + lane) * 16 + e]
//   k   = 16*(e>=8) + 8*(lane>=16) + 2*((e/2)&3) + (e&1)
//   col = tile_n*16 + (lane & 15)
// Each wave's B fragment becomes one contiguous 32B load per lane.
// ---------------------------------------------------------------------------
__global__ __launch_bounds__(256) void pack_b_bf16(const float* __restrict__ B,
                                                   __bf16* __restrict__ out,
                                                   int K, int N)
{
    const int idx = blockIdx.x * 256 + threadIdx.x;      // one per element, K*N total
    if (idx >= K * N) return;
    const int e     = idx & 15;
    const int lane  = (idx >> 4) & 31;
    const int rest  = idx >> 9;
    const int kch   = rest % (K >> 5);
    const int tile  = rest / (K >> 5);
    const int j     = e >> 1;
    const int half  = lane >> 4;
    const int k     = ((j >> 2) << 4) + (half << 3) + ((j & 3) << 1) + (e & 1);
    const int col   = (tile << 4) + (lane & 15);
    out[idx] = (__bf16)B[(kch * 32 + k) * N + col];
}

// ---------------------------------------------------------------------------
// bf16 WMMA GEMM with compile-time K:
//   Cout[M x N] = bias + Cin + A0[MxK0]@B0 + A1[MxK1]@B1
// B0p/B1p are pre-packed fragment-order weights (see pack_b_bf16).
// Wave -> 16x16 tile; block = 8 waves = 16(M) x 128(N); grid = (N/128, M/16).
// EXEC stays all-ones (no divergence) as WMMA requires.
// ---------------------------------------------------------------------------
template <int K0, int K1>
__global__ __launch_bounds__(256) void gemm_bf16_wmma(
    const __bf16* __restrict__ A0, int lda0, const __bf16* __restrict__ B0p,
    const __bf16* __restrict__ A1, int lda1, const __bf16* __restrict__ B1p,
    const float* __restrict__ bias, const float* __restrict__ Cin,
    float* __restrict__ Cout, int N)
{
    const int lane  = threadIdx.x & 31;
    const int wave  = threadIdx.x >> 5;
    const int half  = lane >> 4;
    const int l15   = lane & 15;
    const int row0  = blockIdx.y << 4;                   // M tile base
    const int col0  = (blockIdx.x << 7) + (wave << 4);   // N tile base
    const int tile_n = col0 >> 4;

    v8f acc = {};
    if (Cin) {
#pragma unroll
        for (int r = 0; r < 8; ++r)
            acc[r] = Cin[(row0 + r + (half << 3)) * N + col0 + l15];
    }
    if (bias) {
        const float bv = bias[col0 + l15];
#pragma unroll
        for (int r = 0; r < 8; ++r) acc[r] += bv;
    }

    // --- segment 0: A0 @ B0 ---
    {
        const __bf16* arow = A0 + (size_t)(row0 + l15) * lda0;
        const __bf16* bp   = B0p + ((size_t)tile_n * (K0 >> 5)) * 512 + lane * 16;
#pragma unroll
        for (int kb = 0; kb < K0; kb += 32) {
            const v16bf a = load_a_frag(arow + kb, half);
            const v16bf b = *(const v16bf*)(bp + (size_t)(kb >> 5) * 512);
            acc = __builtin_amdgcn_wmma_f32_16x16x32_bf16(false, a, false, b,
                                                          (short)0, acc, false, false);
        }
    }
    // --- segment 1: A1 @ B1 (recurrent part, optional) ---
    if constexpr (K1 > 0) {
        const __bf16* arow = A1 + (size_t)(row0 + l15) * lda1;
        const __bf16* bp   = B1p + ((size_t)tile_n * (K1 >> 5)) * 512 + lane * 16;
#pragma unroll
        for (int kb = 0; kb < K1; kb += 32) {
            const v16bf a = load_a_frag(arow + kb, half);
            const v16bf b = *(const v16bf*)(bp + (size_t)(kb >> 5) * 512);
            acc = __builtin_amdgcn_wmma_f32_16x16x32_bf16(false, a, false, b,
                                                          (short)0, acc, false, false);
        }
    }

#pragma unroll
    for (int r = 0; r < 8; ++r)
        Cout[(row0 + r + (half << 3)) * N + col0 + l15] = acc[r];
}

// ---------------------------------------------------------------------------
// Encoder pointwise: gates with act = row-softmax over the 512 units.
// One block per batch row, 512 threads. z layout [B, 4*UE] order i,f,g,o.
// ---------------------------------------------------------------------------
__global__ __launch_bounds__(512) void enc_pointwise(
    const float* __restrict__ z, float* __restrict__ c, __bf16* __restrict__ h_bf)
{
    __shared__ float red[512];
    const int b = blockIdx.x;
    const int j = threadIdx.x;
    const float zi = z[b * 4 * UE + j];
    const float zf = z[b * 4 * UE + UE + j];
    const float zg = z[b * 4 * UE + 2 * UE + j];
    const float zo = z[b * 4 * UE + 3 * UE + j];

    // softmax over g row
    red[j] = zg; __syncthreads();
    for (int s = 256; s > 0; s >>= 1) { if (j < s) red[j] = fmaxf(red[j], red[j + s]); __syncthreads(); }
    const float gm = red[0]; __syncthreads();
    const float ge = expf(zg - gm);
    red[j] = ge; __syncthreads();
    for (int s = 256; s > 0; s >>= 1) { if (j < s) red[j] += red[j + s]; __syncthreads(); }
    const float gs = red[0]; __syncthreads();
    const float act_g = ge / gs;

    const float cn = sigmoidf_(zf) * c[b * UE + j] + sigmoidf_(zi) * act_g;
    c[b * UE + j] = cn;

    // softmax over new c row
    red[j] = cn; __syncthreads();
    for (int s = 256; s > 0; s >>= 1) { if (j < s) red[j] = fmaxf(red[j], red[j + s]); __syncthreads(); }
    const float cm = red[0]; __syncthreads();
    const float ce = expf(cn - cm);
    red[j] = ce; __syncthreads();
    for (int s = 256; s > 0; s >>= 1) { if (j < s) red[j] += red[j + s]; __syncthreads(); }
    const float cs = red[0];

    const float h = sigmoidf_(zo) * (ce / cs);
    h_bf[b * UE + j] = (__bf16)h;
}

// ---------------------------------------------------------------------------
// Decoder pointwise: standard tanh LSTM cell, UD=256 units. Also scatters h
// into the output sequence buffer at time index t_store.
// ---------------------------------------------------------------------------
__global__ __launch_bounds__(256) void dec_pointwise(
    const float* __restrict__ z, float* __restrict__ c,
    __bf16* __restrict__ h_bf, float* __restrict__ seq, int t_store)
{
    const int b = blockIdx.x;
    const int j = threadIdx.x;
    const float zi = z[b * 4 * UD + j];
    const float zf = z[b * 4 * UD + UD + j];
    const float zg = z[b * 4 * UD + 2 * UD + j];
    const float zo = z[b * 4 * UD + 3 * UD + j];
    const float cn = sigmoidf_(zf) * c[b * UD + j] + sigmoidf_(zi) * tanhf(zg);
    c[b * UD + j] = cn;
    const float h = sigmoidf_(zo) * tanhf(cn);
    h_bf[b * UD + j] = (__bf16)h;
    seq[(b * TOUT + t_store) * UD + j] = h;
}

// f32 -> bf16 conversion (row-major, used for the A-side input tensor)
__global__ __launch_bounds__(256) void f32_to_bf16(const float* __restrict__ in,
                                                   __bf16* __restrict__ out, int n)
{
    const int i = blockIdx.x * 256 + threadIdx.x;
    if (i < n) out[i] = (__bf16)in[i];
}

// Final TimeDistributed Dense(16): out[b,t,f] = [fwd|bwd] @ Wd + bias
__global__ __launch_bounds__(256) void dense_out_k(
    const float* __restrict__ fwd, const float* __restrict__ bwd,
    const float* __restrict__ Wd, const float* __restrict__ bias,
    float* __restrict__ out)
{
    const int idx = blockIdx.x * 256 + threadIdx.x;   // 256*64*16 total
    const int f = idx & 15;
    const int t = (idx >> 4) & (TOUT - 1);
    const int b = idx >> 10;
    const float* fr = fwd + (b * TOUT + t) * UD;
    const float* br = bwd + (b * TOUT + t) * UD;
    float s = bias[f];
    for (int k = 0; k < UD; ++k) s += fr[k] * Wd[k * 16 + f];
    for (int k = 0; k < UD; ++k) s += br[k] * Wd[(UD + k) * 16 + f];
    out[idx] = s;
}

// ---------------------------------------------------------------------------
extern "C" void kernel_launch(void* const* d_in, const int* in_sizes, int n_in,
                              void* d_out, int out_size, void* d_ws, size_t ws_size,
                              hipStream_t stream)
{
    const float* inputs       = (const float*)d_in[0];
    const float* enc_kernel   = (const float*)d_in[1];
    const float* enc_rec      = (const float*)d_in[2];
    const float* enc_bias     = (const float*)d_in[3];
    const float* dec_f_kernel = (const float*)d_in[4];
    const float* dec_f_rec    = (const float*)d_in[5];
    const float* dec_f_bias   = (const float*)d_in[6];
    const float* dec_b_kernel = (const float*)d_in[7];
    const float* dec_b_rec    = (const float*)d_in[8];
    const float* dec_b_bias   = (const float*)d_in[9];
    const float* dense_kernel = (const float*)d_in[10];
    const float* dense_bias   = (const float*)d_in[11];
    float* out = (float*)d_out;

    char* p = (char*)d_ws;
    auto alloc = [&](size_t bytes) -> char* {
        char* r = p; p += (bytes + 255) & ~(size_t)255; return r;
    };
    __bf16* inputs_bf  = (__bf16*)alloc((size_t)B_ * TIN * FIN * 2);
    __bf16* enc_k_pk   = (__bf16*)alloc((size_t)FIN * 4 * UE * 2);   // packed B
    __bf16* enc_r_pk   = (__bf16*)alloc((size_t)UE * 4 * UE * 2);
    __bf16* dfk_pk     = (__bf16*)alloc((size_t)UE * 4 * UD * 2);
    __bf16* dfr_pk     = (__bf16*)alloc((size_t)UD * 4 * UD * 2);
    __bf16* dbk_pk     = (__bf16*)alloc((size_t)UE * 4 * UD * 2);
    __bf16* dbr_pk     = (__bf16*)alloc((size_t)UD * 4 * UD * 2);
    __bf16* h_enc_bf   = (__bf16*)alloc((size_t)B_ * UE * 2);
    float*  c_enc      = (float*) alloc((size_t)B_ * UE * 4);
    float*  z_enc      = (float*) alloc((size_t)B_ * 4 * UE * 4);
    float*  xWf        = (float*) alloc((size_t)B_ * 4 * UD * 4);
    float*  xWb        = (float*) alloc((size_t)B_ * 4 * UD * 4);
    __bf16* h_f_bf     = (__bf16*)alloc((size_t)B_ * UD * 2);
    float*  c_f        = (float*) alloc((size_t)B_ * UD * 4);
    __bf16* h_b_bf     = (__bf16*)alloc((size_t)B_ * UD * 2);
    float*  c_b        = (float*) alloc((size_t)B_ * UD * 4);
    float*  z_f        = (float*) alloc((size_t)B_ * 4 * UD * 4);
    float*  z_b        = (float*) alloc((size_t)B_ * 4 * UD * 4);
    float*  fwd_seq    = (float*) alloc((size_t)B_ * TOUT * UD * 4);
    float*  bwd_seq    = (float*) alloc((size_t)B_ * TOUT * UD * 4);

    // ---- one-time conversions / weight packing ----
    {
        const int n = B_ * TIN * FIN;
        f32_to_bf16<<<(n + 255) / 256, 256, 0, stream>>>(inputs, inputs_bf, n);
    }
    auto pack = [&](const float* src, __bf16* dst, int K, int N) {
        const int n = K * N;
        pack_b_bf16<<<(n + 255) / 256, 256, 0, stream>>>(src, dst, K, N);
    };
    pack(enc_kernel,   enc_k_pk, FIN, 4 * UE);
    pack(enc_rec,      enc_r_pk, UE,  4 * UE);
    pack(dec_f_kernel, dfk_pk,   UE,  4 * UD);
    pack(dec_f_rec,    dfr_pk,   UD,  4 * UD);
    pack(dec_b_kernel, dbk_pk,   UE,  4 * UD);
    pack(dec_b_rec,    dbr_pk,   UD,  4 * UD);

    // ---- zero initial states (graph-capturable) ----
    hipMemsetAsync(h_enc_bf, 0, (size_t)B_ * UE * 2, stream);
    hipMemsetAsync(c_enc,    0, (size_t)B_ * UE * 4, stream);
    hipMemsetAsync(h_f_bf,   0, (size_t)B_ * UD * 2, stream);
    hipMemsetAsync(c_f,      0, (size_t)B_ * UD * 4, stream);
    hipMemsetAsync(h_b_bf,   0, (size_t)B_ * UD * 2, stream);
    hipMemsetAsync(c_b,      0, (size_t)B_ * UD * 4, stream);

    // ---- encoder: 256 chained steps; z = x_t@W + h@U + b fused in one WMMA GEMM ----
    {
        const dim3 grid(4 * UE / 128, B_ / 16);   // (16,16)
        for (int t = 0; t < TIN; ++t) {
            gemm_bf16_wmma<FIN, UE><<<grid, 256, 0, stream>>>(
                inputs_bf + (size_t)t * FIN, TIN * FIN, enc_k_pk,
                h_enc_bf, UE, enc_r_pk,
                enc_bias, nullptr, z_enc, 4 * UE);
            enc_pointwise<<<B_, 512, 0, stream>>>(z_enc, c_enc, h_enc_bf);
        }
    }

    // ---- decoder input projections: RepeatVector => time-invariant xW ----
    {
        const dim3 grid(4 * UD / 128, B_ / 16);   // (8,16)
        gemm_bf16_wmma<UE, 0><<<grid, 256, 0, stream>>>(
            h_enc_bf, UE, dfk_pk, nullptr, 0, nullptr,
            dec_f_bias, nullptr, xWf, 4 * UD);
        gemm_bf16_wmma<UE, 0><<<grid, 256, 0, stream>>>(
            h_enc_bf, UE, dbk_pk, nullptr, 0, nullptr,
            dec_b_bias, nullptr, xWb, 4 * UD);
    }

    // ---- bidirectional decoder: 64 chained steps each direction ----
    {
        const dim3 grid(4 * UD / 128, B_ / 16);   // (8,16)
        for (int s = 0; s < TOUT; ++s) {
            // forward: store at t = s
            gemm_bf16_wmma<UD, 0><<<grid, 256, 0, stream>>>(
                h_f_bf, UD, dfr_pk, nullptr, 0, nullptr,
                nullptr, xWf, z_f, 4 * UD);
            dec_pointwise<<<B_, 256, 0, stream>>>(z_f, c_f, h_f_bf, fwd_seq, s);
            // backward: input sequence is constant in time, so reversal only
            // permutes the output index: store at t = TOUT-1-s
            gemm_bf16_wmma<UD, 0><<<grid, 256, 0, stream>>>(
                h_b_bf, UD, dbr_pk, nullptr, 0, nullptr,
                nullptr, xWb, z_b, 4 * UD);
            dec_pointwise<<<B_, 256, 0, stream>>>(z_b, c_b, h_b_bf, bwd_seq, TOUT - 1 - s);
        }
    }

    // ---- TimeDistributed Dense(16) ----
    dense_out_k<<<(B_ * TOUT * 16) / 256, 256, 0, stream>>>(
        fwd_seq, bwd_seq, dense_kernel, dense_bias, out);
}
